// Attention_80530636800482
// MI455X (gfx1250) — matrix operand order repaired and checked by reading, not compiled, not run on hardware
//
#include <hip/hip_runtime.h>

typedef __bf16 bf16_t;
typedef __attribute__((ext_vector_type(16))) __bf16 v16bf;
typedef __attribute__((ext_vector_type(8)))  __bf16 v8bf;
typedef __attribute__((ext_vector_type(8)))  float  v8f;

#define D_MODEL 1536
#define NUM_HEADS 16
#define QUERY_GROUPS 4
#define HEAD_DIM 96
#define SEQ_N 2048
#define BATCH 2
#define ROWS (BATCH*SEQ_N)      /* 4096 */
#define QKV_OUT 2304

// ---------- fragment helpers (layouts per CDNA5 ISA 7.12.2, wave32) ----------
static __device__ __forceinline__ v16bf load2(const bf16_t* p0, const bf16_t* p1) {
    v8bf lo = *reinterpret_cast<const v8bf*>(p0);   // 16B -> *_load_b128
    v8bf hi = *reinterpret_cast<const v8bf*>(p1);
    return __builtin_shufflevector(lo, hi, 0,1,2,3,4,5,6,7,8,9,10,11,12,13,14,15);
}
static __device__ __forceinline__ v8f wmma_bf(v16bf a, v16bf b, v8f c) {
    return __builtin_amdgcn_wmma_f32_16x16x32_bf16(false, a, false, b, (short)0, c, false, false);
}

// ---------- fp32 -> bf16 cast ----------
__global__ void to_bf16_k(const float* __restrict__ in, bf16_t* __restrict__ out, int n) {
    int i = blockIdx.x * 256 + threadIdx.x;
    if (i < n) out[i] = (bf16_t)in[i];
}

// ---------- generic bf16 GEMM: C[M,N] = A[M,K] @ Bm[N,K]^T, fp32 out ----------
// one wave computes a 32x64 tile (2 M-subtiles x 4 N-subtiles):
// each B fragment feeds 2 WMMAs, each A fragment feeds 4 -> 8 WMMAs / 12 b128 loads.
// M % 32 == 0, N % 64 == 0, K % 32 == 0
__global__ __launch_bounds__(256) void gemm_bf16_k(const bf16_t* __restrict__ A,
                                                   const bf16_t* __restrict__ Bm,
                                                   float* __restrict__ C,
                                                   int M, int N, int K)
{
    int wid  = (int)((blockIdx.x * blockDim.x + threadIdx.x) >> 5);
    int lane = threadIdx.x & 31;
    int r = lane & 15, half = lane >> 4;
    int ntiles = N >> 6;
    int tm = wid / ntiles, tn = wid - tm * ntiles;
    int m0 = tm << 5, n0 = tn << 6;
    if (m0 >= M) return;

    v8f acc[8];
#pragma unroll
    for (int t = 0; t < 8; ++t)
#pragma unroll
        for (int i = 0; i < 8; ++i) acc[t][i] = 0.0f;

    const bf16_t* arow0 = A  + (size_t)(m0 +      r) * K + half * 8;
    const bf16_t* arow1 = A  + (size_t)(m0 + 16 + r) * K + half * 8;
    const bf16_t* b0    = Bm + (size_t)(n0 +  0 + r) * K + half * 8;
    const bf16_t* b1    = Bm + (size_t)(n0 + 16 + r) * K + half * 8;
    const bf16_t* b2    = Bm + (size_t)(n0 + 32 + r) * K + half * 8;
    const bf16_t* b3    = Bm + (size_t)(n0 + 48 + r) * K + half * 8;

    for (int kk = 0; kk < K; kk += 32) {
        v16bf af0 = load2(arow0 + kk, arow0 + kk + 16);
        v16bf af1 = load2(arow1 + kk, arow1 + kk + 16);
        v16bf bf0 = load2(b0 + kk, b0 + kk + 16);
        acc[0] = wmma_bf(af0, bf0, acc[0]);
        acc[4] = wmma_bf(af1, bf0, acc[4]);
        v16bf bf1 = load2(b1 + kk, b1 + kk + 16);
        acc[1] = wmma_bf(af0, bf1, acc[1]);
        acc[5] = wmma_bf(af1, bf1, acc[5]);
        v16bf bf2 = load2(b2 + kk, b2 + kk + 16);
        acc[2] = wmma_bf(af0, bf2, acc[2]);
        acc[6] = wmma_bf(af1, bf2, acc[6]);
        v16bf bf3 = load2(b3 + kk, b3 + kk + 16);
        acc[3] = wmma_bf(af0, bf3, acc[3]);
        acc[7] = wmma_bf(af1, bf3, acc[7]);
    }
#pragma unroll
    for (int mt = 0; mt < 2; ++mt) {
#pragma unroll
        for (int i = 0; i < 8; ++i) {
            size_t ro = (size_t)(m0 + mt * 16 + 8 * half + i) * N + n0;
            C[ro +  0 + r] = acc[4 * mt + 0][i];
            C[ro + 16 + r] = acc[4 * mt + 1][i];
            C[ro + 32 + r] = acc[4 * mt + 2][i];
            C[ro + 48 + r] = acc[4 * mt + 3][i];
        }
    }
}

// ---------- RoPE on q & k slices of fp32 qkv; emit bf16 q[b,h,n,96], k[b,g,n,96] ----------
__global__ void rope_qk_k(const float* __restrict__ qkv,
                          bf16_t* __restrict__ qb, bf16_t* __restrict__ kb)
{
    int idx = blockIdx.x * 256 + threadIdx.x;            // ROWS*20*48 threads
    if (idx >= ROWS * 20 * 48) return;
    int p    = idx & 15;                                 // rotation pair (P=16)
    int axis = (idx >> 4) % 3;
    int hh   = (idx / 48) % 20;                          // 0..15 q heads, 16..19 k groups
    int row  = idx / 960;
    int b = row >> 11, n = row & 2047;
    int pos = (axis == 0) ? (n >> 8) : (axis == 1) ? ((n >> 4) & 15) : (n & 15);
    float freq = __powf(10000.0f, -(float)p * (1.0f / 16.0f));   // theta^(-2p/32)
    float s, c; __sincosf((float)pos * freq, &s, &c);
    int d = axis * 32 + 2 * p;
    int col = (hh < 16) ? hh * 96 + d : 1536 + (hh - 16) * 96 + d;
    float x0 = qkv[(size_t)row * QKV_OUT + col];
    float x1 = qkv[(size_t)row * QKV_OUT + col + 1];
    float y0 = x0 * c - x1 * s;
    float y1 = x0 * s + x1 * c;
    if (hh < 16) {
        size_t o = ((size_t)((b * NUM_HEADS + hh) * SEQ_N + n)) * HEAD_DIM + d;
        qb[o] = (bf16_t)y0; qb[o + 1] = (bf16_t)y1;
    } else {
        size_t o = ((size_t)((b * QUERY_GROUPS + (hh - 16)) * SEQ_N + n)) * HEAD_DIM + d;
        kb[o] = (bf16_t)y0; kb[o + 1] = (bf16_t)y1;
    }
}

// ---------- pack V transposed: vtb[b,g,d,n] = qkv[row, 1920 + g*96 + d] ----------
__global__ void v_pack_k(const float* __restrict__ qkv, bf16_t* __restrict__ vtb)
{
    int idx = blockIdx.x * 256 + threadIdx.x;            // BATCH*G*96*2048 threads
    if (idx >= BATCH * QUERY_GROUPS * HEAD_DIM * SEQ_N) return;
    int n = idx & 2047;
    int rest = idx >> 11;
    int d = rest % HEAD_DIM;
    int rest2 = rest / HEAD_DIM;
    int g = rest2 & 3, b = rest2 >> 2;
    float v = qkv[(size_t)(b * SEQ_N + n) * QKV_OUT + 1920 + g * 96 + d];
    vtb[idx] = (bf16_t)v;
}

// ---------- flash attention: 4 waves/block, one 16-query tile per wave ----------
__global__ __launch_bounds__(128) void attn_k(const bf16_t* __restrict__ qb,
                                              const bf16_t* __restrict__ kb,
                                              const bf16_t* __restrict__ vtb,
                                              bf16_t* __restrict__ attb)
{
    __shared__ __align__(16) float  Sb[4][16][32];
    __shared__ __align__(16) bf16_t Pb[4][16][32];
    __shared__ float Cb[4][16];

    int w = threadIdx.x >> 5;
    int lane = threadIdx.x & 31;
    int r = lane & 15, half = lane >> 4;
    int bid = blockIdx.x;                 // = ((b*16 + h)*32 + qt)
    int qt = bid & 31;
    int h  = (bid >> 5) & 15;
    int b  = bid >> 9;
    int n0 = qt * 64 + w * 16;
    int g  = h >> 2;                      // HEADS_PER_GROUP = 4

    const bf16_t* qrow = qb + ((size_t)((b * NUM_HEADS + h) * SEQ_N + n0 + r)) * HEAD_DIM;
    v16bf qf[3];
#pragma unroll
    for (int kk = 0; kk < 3; ++kk) {
        const bf16_t* p = qrow + kk * 32 + half * 8;
        qf[kk] = load2(p, p + 16);
    }
    const bf16_t* kbase = kb  + ((size_t)(b * QUERY_GROUPS + g)) * SEQ_N * HEAD_DIM;
    const bf16_t* vbase = vtb + ((size_t)(b * QUERY_GROUPS + g)) * HEAD_DIM * SEQ_N;

    v8f o[6];
#pragma unroll
    for (int t = 0; t < 6; ++t)
#pragma unroll
        for (int i = 0; i < 8; ++i) o[t][i] = 0.0f;

    float m_i = -1e30f, l_i = 0.0f;
    const float scale = 0.1020620726159658f;   // 1/sqrt(96)

    for (int j0 = 0; j0 < SEQ_N; j0 += 32) {
        // ---- S = Q @ K^T (16x32 tile, hd=96 => 3 K-chunks) ----
        v8f s0 = {}, s1 = {};
#pragma unroll
        for (int kk = 0; kk < 3; ++kk) {
            const bf16_t* p0 = kbase + (size_t)(j0 +      r) * HEAD_DIM + kk * 32 + half * 8;
            const bf16_t* p1 = kbase + (size_t)(j0 + 16 + r) * HEAD_DIM + kk * 32 + half * 8;
            s0 = wmma_bf(qf[kk], load2(p0, p0 + 16), s0);
            s1 = wmma_bf(qf[kk], load2(p1, p1 + 16), s1);
        }
#pragma unroll
        for (int i = 0; i < 8; ++i) {            // D layout: row = 8*half+i, col = r
            Sb[w][8 * half + i][r]      = s0[i] * scale;
            Sb[w][8 * half + i][16 + r] = s1[i] * scale;
        }
        __syncthreads();
        // ---- online softmax: lane < 16 owns query row `lane` ----
        if (lane < 16) {
            float rm = m_i;
#pragma unroll
            for (int j = 0; j < 32; ++j) rm = fmaxf(rm, Sb[w][lane][j]);
            float corr = __expf(m_i - rm);
            float rs = 0.0f;
#pragma unroll
            for (int j = 0; j < 32; ++j) {
                float e = __expf(Sb[w][lane][j] - rm);
                rs += e;
                Pb[w][lane][j] = (bf16_t)e;
            }
            l_i = l_i * corr + rs;
            m_i = rm;
            Cb[w][lane] = corr;
        }
        __syncthreads();
        // ---- rescale accumulators by per-row correction ----
        float cc[8];
#pragma unroll
        for (int i = 0; i < 8; ++i) cc[i] = Cb[w][8 * half + i];
#pragma unroll
        for (int t = 0; t < 6; ++t)
#pragma unroll
            for (int i = 0; i < 8; ++i) o[t][i] *= cc[i];
        // ---- O += P @ V (LDS round-trip gave us P in A-fragment layout) ----
        const bf16_t* pp = &Pb[w][r][half * 8];
        v16bf pf = load2(pp, pp + 16);
#pragma unroll
        for (int t = 0; t < 6; ++t) {
            const bf16_t* vp = vbase + (size_t)(t * 16 + r) * SEQ_N + j0 + half * 8;
            o[t] = wmma_bf(pf, load2(vp, vp + 16), o[t]);
        }
    }

    // ---- finalize: divide by l, write bf16 att[b, n, h*96 + d] ----
    __syncthreads();
    if (lane < 16) Cb[w][lane] = 1.0f / l_i;
    __syncthreads();
    float inv[8];
#pragma unroll
    for (int i = 0; i < 8; ++i) inv[i] = Cb[w][8 * half + i];
    bf16_t* ob = attb + ((size_t)(b * SEQ_N + n0)) * D_MODEL + h * HEAD_DIM;
#pragma unroll
    for (int t = 0; t < 6; ++t)
#pragma unroll
        for (int i = 0; i < 8; ++i)
            ob[(size_t)(8 * half + i) * D_MODEL + t * 16 + r] = (bf16_t)(o[t][i] * inv[i]);
}

// ---------------------------------------------------------------------------
extern "C" void kernel_launch(void* const* d_in, const int* in_sizes, int n_in,
                              void* d_out, int out_size, void* d_ws, size_t ws_size,
                              hipStream_t stream)
{
    (void)in_sizes; (void)n_in; (void)out_size; (void)ws_size;
    const float* x     = (const float*)d_in[0];
    const float* w_qkv = (const float*)d_in[1];
    const float* w_o   = (const float*)d_in[2];
    float* out = (float*)d_out;

    char* ws = (char*)d_ws;
    size_t off = 0;
    auto alloc = [&](size_t bytes) -> void* {
        void* p = ws + off;
        off = (off + bytes + 255) & ~(size_t)255;
        return p;
    };
    bf16_t* xb  = (bf16_t*)alloc((size_t)ROWS * D_MODEL * 2);
    bf16_t* wqb = (bf16_t*)alloc((size_t)QKV_OUT * D_MODEL * 2);
    bf16_t* wob = (bf16_t*)alloc((size_t)D_MODEL * D_MODEL * 2);
    float*  qkv = (float*) alloc((size_t)ROWS * QKV_OUT * 4);
    bf16_t* qb  = (bf16_t*)alloc((size_t)BATCH * NUM_HEADS * SEQ_N * HEAD_DIM * 2);
    bf16_t* kb  = (bf16_t*)alloc((size_t)BATCH * QUERY_GROUPS * SEQ_N * HEAD_DIM * 2);
    bf16_t* vtb = (bf16_t*)alloc((size_t)BATCH * QUERY_GROUPS * HEAD_DIM * SEQ_N * 2);
    bf16_t* attb = xb;   // xb is dead after the QKV GEMM; reuse (same size)

    int nx = ROWS * D_MODEL;
    to_bf16_k<<<nx / 256, 256, 0, stream>>>(x, xb, nx);
    int nq = QKV_OUT * D_MODEL;
    to_bf16_k<<<nq / 256, 256, 0, stream>>>(w_qkv, wqb, nq);
    int no = D_MODEL * D_MODEL;
    to_bf16_k<<<no / 256, 256, 0, stream>>>(w_o, wob, no);

    // qkv = x @ w_qkv^T : [4096,2304]; (4096/32)*(2304/64) = 4608 waves / 8 per block
    gemm_bf16_k<<<(ROWS / 32) * (QKV_OUT / 64) / 8, 256, 0, stream>>>(
        xb, wqb, qkv, ROWS, QKV_OUT, D_MODEL);

    rope_qk_k<<<(ROWS * 20 * 48) / 256, 256, 0, stream>>>(qkv, qb, kb);
    v_pack_k<<<(BATCH * QUERY_GROUPS * HEAD_DIM * SEQ_N) / 256, 256, 0, stream>>>(qkv, vtb);

    attn_k<<<BATCH * NUM_HEADS * (SEQ_N / 64), 128, 0, stream>>>(qb, kb, vtb, attb);

    // out = att @ w_o^T : [4096,1536]; (4096/32)*(1536/64) = 3072 waves / 8 per block
    gemm_bf16_k<<<(ROWS / 32) * (D_MODEL / 64) / 8, 256, 0, stream>>>(
        attb, wob, out, ROWS, D_MODEL, D_MODEL);
}
